// Attention_1142461301310
// MI455X (gfx1250) — matrix operand order repaired and checked by reading, not compiled, not run on hardware
//
#include <hip/hip_runtime.h>
#include <hip/hip_bf16.h>

#define HIDDEN 512
#define TWO_H  1024
#define BATCH  64
#define SEQ    1024

typedef __attribute__((ext_vector_type(16))) _Float16 v16h;
typedef __attribute__((ext_vector_type(8)))  float    v8f;

// Native CDNA5 v_tanh_f32 when available (single TRANS op instead of ~20-op
// libm polynomial; keeps the 256MB scores stream memory-bound, not VALU-bound).
#if __has_builtin(__builtin_amdgcn_tanhf)
  #define FAST_TANH(x) __builtin_amdgcn_tanhf(x)
#else
  #define FAST_TANH(x) tanhf(x)
#endif

// ---------------------------------------------------------------------------
// Kernel 1: base[64,1024] = hidden[64,1024] @ W_feat[1024,1024] + b_feat + b_cov
// One wave32 per 16x16 output tile, v_wmma_f32_16x16x32_f16, K-loop step 32.
// Fragment packing: contiguous K order k = 16*(lane/16) + s for both A 16x32 and B 32x16, f32 C/D.
// b_cov is folded into the epilogue so the hot scores pass reads one less stream.
// ---------------------------------------------------------------------------
__global__ void dec_gemm_wmma(const float* __restrict__ hidden,
                              const float* __restrict__ W_feat,
                              const float* __restrict__ b_feat,
                              const float* __restrict__ b_cov,
                              float* __restrict__ dec) {
    const int lane  = threadIdx.x & 31;
    const int wave  = threadIdx.x >> 5;
    const int tileN = blockIdx.x * 4 + wave;   // 64 N-tiles, 4 waves/block
    const int tileM = blockIdx.y;              // 4 M-tiles

    const int m     = lane & 15;
    const int arow  = tileM * 16 + m;           // A row (M) for this lane
    const int koff  = (lane < 16) ? 0 : 8;      // unused; A and B both index K via kbase
    const int n     = tileN * 16 + (lane & 15); // B/C/D column (N) for this lane
    const int kbase = (lane < 16) ? 0 : 16;     // B: K base per half-wave

    v8f c = {};
    const float* __restrict__ Arow = hidden + (size_t)arow * TWO_H;

    for (int k0 = 0; k0 < TWO_H; k0 += 32) {
        v16h a, b;
        // A fragment: halves 0..15 = K=k0+kbase+0..15 (same K order as B)
        #pragma unroll
        for (int j = 0; j < 8; ++j) {
            a[j]     = (_Float16)Arow[k0 + kbase + j];
            a[j + 8] = (_Float16)Arow[k0 + kbase + 8 + j];
        }
        // B fragment: halves j = B[k0 + kbase + j, n], column n, strided by row pitch
        const float* __restrict__ Bcol = W_feat + (size_t)(k0 + kbase) * TWO_H + n;
        #pragma unroll
        for (int j = 0; j < 16; ++j)
            b[j] = (_Float16)Bcol[(size_t)j * TWO_H];

        c = __builtin_amdgcn_wmma_f32_16x16x32_f16(
                /*neg_a=*/false, a, /*neg_b=*/false, b,
                /*c_mod=*/(short)0, c, /*reuse_a=*/false, /*reuse_b=*/false);
    }

    // D layout: VGPR r -> M = r (lanes 0-15) or r+8 (lanes 16-31), N = lane&15
    const int   mrow0 = tileM * 16 + ((lane < 16) ? 0 : 8);
    const float bias  = b_feat[n] + b_cov[n];
    #pragma unroll
    for (int r = 0; r < 8; ++r)
        dec[(size_t)(mrow0 + r) * TWO_H + n] = c[r] + bias;
}

// ---------------------------------------------------------------------------
// Kernel 2: scores[s,b] = tanh(ef[s,b,:] + base[b,:] + cov[s,b]*W_cov) . W_att + b_att
// One wave per row (s*B+b); 8 x float4 coalesced loads per lane; wave32 shuffle
// reduce. This is the 256 MB encoder_features stream — fused, read exactly once.
// base/W_cov/W_att are 4-16KB and stay resident in WGP$/L2.
// ---------------------------------------------------------------------------
__global__ void scores_kernel(const float* __restrict__ ef,
                              const float* __restrict__ base,
                              const float* __restrict__ coverage,
                              const float* __restrict__ W_cov,
                              const float* __restrict__ W_att,
                              const float* __restrict__ b_att,
                              float* __restrict__ scores) {
    const int lane = threadIdx.x & 31;
    const int wid  = threadIdx.x >> 5;
    const int row  = blockIdx.x * (blockDim.x >> 5) + wid;  // row = s*BATCH + b
    if (row >= SEQ * BATCH) return;
    const int   bcol = row & (BATCH - 1);
    const float cov  = coverage[row];

    const float* __restrict__ e  = ef   + (size_t)row  * TWO_H;
    const float* __restrict__ dv = base + (size_t)bcol * TWO_H;

    float acc = 0.f;
    #pragma unroll
    for (int i = 0; i < 8; ++i) {
        const int h = (i * 32 + lane) * 4;
        float4 ev = *(const float4*)(e  + h);
        float4 dd = *(const float4*)(dv + h);
        float4 wc = *(const float4*)(W_cov + h);
        float4 wa = *(const float4*)(W_att + h);   // W_att is [2H,1] -> contiguous
        acc += FAST_TANH(ev.x + dd.x + cov * wc.x) * wa.x;
        acc += FAST_TANH(ev.y + dd.y + cov * wc.y) * wa.y;
        acc += FAST_TANH(ev.z + dd.z + cov * wc.z) * wa.z;
        acc += FAST_TANH(ev.w + dd.w + cov * wc.w) * wa.w;
    }
    #pragma unroll
    for (int off = 16; off > 0; off >>= 1)
        acc += __shfl_xor(acc, off, 32);
    if (lane == 0) scores[row] = acc + b_att[0];
}

// ---------------------------------------------------------------------------
// Kernel 3: masked softmax over seq (per batch column) + renorm + coverage.
// a = exp(s-max)*mask / sum(exp(s-max)*mask) ; new_cov = cov + a
// One block per batch column, 1024 threads, LDS tree reductions.
// ---------------------------------------------------------------------------
__global__ void softmax_kernel(const float* __restrict__ scores,
                               const float* __restrict__ mask,
                               const float* __restrict__ coverage,
                               float* __restrict__ a_out,
                               float* __restrict__ cov_out) {
    __shared__ float red[SEQ];
    const int b = blockIdx.x;
    const int s = threadIdx.x;
    const int idx = s * BATCH + b;

    const float sc = scores[idx];
    red[s] = sc;
    __syncthreads();
    #pragma unroll
    for (int off = SEQ / 2; off > 0; off >>= 1) {
        if (s < off) red[s] = fmaxf(red[s], red[s + off]);
        __syncthreads();
    }
    const float mx = red[0];
    __syncthreads();

    const float e = __expf(sc - mx) * mask[idx];
    red[s] = e;
    __syncthreads();
    #pragma unroll
    for (int off = SEQ / 2; off > 0; off >>= 1) {
        if (s < off) red[s] += red[s + off];
        __syncthreads();
    }
    const float a = e / red[0];
    a_out[idx]   = a;
    cov_out[idx] = coverage[idx] + a;
}

// ---------------------------------------------------------------------------
// Kernel 4: context partials. context[b,h] = sum_s a[s,b] * eo[b,s,h]
// Block = (b, s-segment of 128). a-segment cached in LDS; float4 over h.
// 512 blocks / 4096 waves keep enough loads in flight to saturate HBM.
// This is the 256 MB encoder_out stream — read exactly once.
// ---------------------------------------------------------------------------
#define SSEG 8
#define SLEN (SEQ / SSEG)

__global__ void context_partial_kernel(const float* __restrict__ eo,
                                       const float* __restrict__ a,
                                       float* __restrict__ partial) {
    __shared__ float as[SLEN];
    const int b  = blockIdx.x;
    const int z  = blockIdx.y;          // s-segment
    const int t  = threadIdx.x;         // 256 threads, each owns 4 h values
    const int s0 = z * SLEN;

    if (t < SLEN) as[t] = a[(size_t)(s0 + t) * BATCH + b];
    __syncthreads();

    const int h = t * 4;
    const float* __restrict__ ep = eo + (size_t)b * SEQ * TWO_H + (size_t)s0 * TWO_H + h;
    float4 acc = make_float4(0.f, 0.f, 0.f, 0.f);
    #pragma unroll 4
    for (int s = 0; s < SLEN; ++s) {
        const float  w  = as[s];
        const float4 ev = *(const float4*)(ep + (size_t)s * TWO_H);
        acc.x += w * ev.x; acc.y += w * ev.y; acc.z += w * ev.z; acc.w += w * ev.w;
    }
    *(float4*)(partial + ((size_t)z * BATCH + b) * TWO_H + h) = acc;
}

__global__ void context_reduce_kernel(const float* __restrict__ partial,
                                      float* __restrict__ ctx) {
    const int i = blockIdx.x * blockDim.x + threadIdx.x;   // over B*2H
    float acc = 0.f;
    #pragma unroll
    for (int z = 0; z < SSEG; ++z)
        acc += partial[(size_t)z * BATCH * TWO_H + i];
    ctx[i] = acc;
}

// ---------------------------------------------------------------------------
extern "C" void kernel_launch(void* const* d_in, const int* in_sizes, int n_in,
                              void* d_out, int out_size, void* d_ws, size_t ws_size,
                              hipStream_t stream) {
    const float* hidden    = (const float*)d_in[0];   // [64, 1024]
    const float* enc_out   = (const float*)d_in[1];   // [64, 1024, 1024]
    const float* enc_feat  = (const float*)d_in[2];   // [65536, 1024]
    const float* enc_mask  = (const float*)d_in[3];   // [1024, 64]
    const float* coverage  = (const float*)d_in[4];   // [1024, 64]
    const float* W_feat    = (const float*)d_in[5];   // [1024, 1024]
    const float* b_feat    = (const float*)d_in[6];   // [1024]
    const float* W_cov     = (const float*)d_in[7];   // [1, 1024]
    const float* b_cov     = (const float*)d_in[8];   // [1024]
    const float* W_att     = (const float*)d_in[9];   // [1024, 1]
    const float* b_att     = (const float*)d_in[10];  // [1]

    float* out = (float*)d_out;
    float* ctx_out = out;                              // [64, 1024]
    float* a_out   = out + BATCH * TWO_H;              // [1024, 64]
    float* cov_out = a_out + SEQ * BATCH;              // [1024, 64]

    float* ws      = (float*)d_ws;
    float* base    = ws;                               // 65536 floats
    float* scores  = ws + BATCH * TWO_H;               // 65536 floats
    float* partial = scores + SEQ * BATCH;             // SSEG*65536 floats

    // 1) base = hidden @ W_feat + b_feat + b_cov   (WMMA f16->f32)
    dec_gemm_wmma<<<dim3(16, 4), 128, 0, stream>>>(hidden, W_feat, b_feat, b_cov, base);

    // 2) fused attention scores over the encoder_features stream
    scores_kernel<<<dim3((SEQ * BATCH) / 8), 256, 0, stream>>>(
        enc_feat, base, coverage, W_cov, W_att, b_att, scores);

    // 3) masked softmax + renorm + coverage update (writes a, new_coverage)
    softmax_kernel<<<dim3(BATCH), SEQ, 0, stream>>>(
        scores, enc_mask, coverage, a_out, cov_out);

    // 4) context = einsum("sb,bsh->bh") over the encoder_out stream
    context_partial_kernel<<<dim3(BATCH, SSEG), 256, 0, stream>>>(
        enc_out, a_out, partial);
    context_reduce_kernel<<<dim3((BATCH * TWO_H) / 256), 256, 0, stream>>>(
        partial, ctx_out);
}
